// OptimizedFluxIPAdapterAttnProcessor_42949672960458
// MI455X (gfx1250) — matrix operand order repaired and checked by reading, not compiled, not run on hardware
//
#include <hip/hip_runtime.h>

// ---------------- problem constants ----------------
#define HS    3072
#define HEADS 24
#define HD    128
#define S_IMG 2048
#define S_TXT 512
#define STOT  2560
#define NTOK  4
#define CA    2048
#define RSQRT_D 0.08838834764831845f

typedef __attribute__((ext_vector_type(16))) __bf16 v16bf;
typedef __attribute__((ext_vector_type(8)))  float  v8f;

static __device__ __forceinline__ v8f wmma_bf16(v16bf a, v16bf b, v8f c) {
  // D = A(16x32 bf16) * B(32x16 bf16) + C(16x16 f32)
  return __builtin_amdgcn_wmma_f32_16x16x32_bf16(
      false, a, false, b, (short)0, c, false, false);
}

// ------- CDNA5 async global->LDS copy (ASYNCcnt path), with fallback -------
#if defined(__AMDGCN__) && __has_builtin(__builtin_amdgcn_global_load_async_to_lds_b128)
#define HAVE_ASYNC_G2L 1
typedef int v4i_t __attribute__((ext_vector_type(4)));
typedef __attribute__((address_space(1))) v4i_t* gptr_v4i;   // global int4*
typedef __attribute__((address_space(3))) v4i_t* lptr_v4i;   // LDS int4*
static __device__ __forceinline__ void async_g2l_b128(const void* g, void* l) {
  __builtin_amdgcn_global_load_async_to_lds_b128((gptr_v4i)g, (lptr_v4i)l, 0, 0);
}
static __device__ __forceinline__ void wait_async0() {
#if __has_builtin(__builtin_amdgcn_s_wait_asynccnt)
  __builtin_amdgcn_s_wait_asynccnt(0);
#else
  asm volatile("s_wait_asynccnt 0x0" ::: "memory");
#endif
}
#else
#define HAVE_ASYNC_G2L 0
#endif

// =====================================================================
// Wave-level GEMM:  C[M,N] = A[M,K] @ W[N,K]^T + bias[N]
// one wave (32 lanes) computes a 32x64 tile; 2x4 WMMA fragments
// (A fragment reused 4x per K-step -> 8 wmma per 32-deep K-step).
// A may be float or __bf16 (converted to bf16 on the fly).
// =====================================================================
template <typename AT>
__global__ void gemm32(const AT* __restrict__ A, const float* __restrict__ W,
                       const float* __restrict__ bias, float* __restrict__ C,
                       int M, int N, int K, int lda, int ldc) {
  const int lane = threadIdx.x & 31;
  const int half = lane >> 4;       // 0: lanes 0-15, 1: lanes 16-31
  const int r    = lane & 15;
  const int n0 = blockIdx.x * 64;
  const int m0 = blockIdx.y * 32;

  v8f acc[2][4] = {};

  for (int k0 = 0; k0 < K; k0 += 32) {
    v16bf a[2], b[4];
#pragma unroll
    for (int i = 0; i < 2; ++i) {
      int m = m0 + i * 16 + r;
      if (m >= M) m = M - 1;                       // clamp (stores are guarded)
      const AT* arow = A + (size_t)m * lda + k0 + half * 16;
#pragma unroll
      for (int j = 0; j < 8; ++j) {
        a[i][j]     = (__bf16)(float)arow[j];       // K chunk  {c0..c0+7}
        a[i][8 + j] = (__bf16)(float)arow[8 + j];   // K chunk  {c0+8..c0+15}
      }
    }
#pragma unroll
    for (int jt = 0; jt < 4; ++jt) {
      const int n = n0 + jt * 16 + r;
      const float* wrow = W + (size_t)n * K + k0 + half * 16;
      __builtin_prefetch(wrow + 32, 0, 1);          // global_prefetch next K tile
#pragma unroll
      for (int j = 0; j < 16; ++j) b[jt][j] = (__bf16)wrow[j];
    }
#pragma unroll
    for (int i = 0; i < 2; ++i)
#pragma unroll
      for (int jt = 0; jt < 4; ++jt)
        acc[i][jt] = wmma_bf16(a[i], b[jt], acc[i][jt]);
  }

  // C/D layout: vgpr rr, lane -> (m = rr + half*8, n = lane&15) within 16x16 tile
#pragma unroll
  for (int i = 0; i < 2; ++i) {
    const int mbase = m0 + i * 16 + half * 8;
#pragma unroll
    for (int jt = 0; jt < 4; ++jt) {
      const int n = n0 + jt * 16 + r;
      const float bv = bias ? bias[n] : 0.0f;
#pragma unroll
      for (int rr = 0; rr < 8; ++rr) {
        const int mm = mbase + rr;
        if (mm < M) C[(size_t)mm * ldc + n] = acc[i][jt][rr] + bv;
      }
    }
  }
}

// =====================================================================
// Fused RMSNorm (+ optional interleaved RoPE) + scale; emits bf16 [H][S][D]
// grid: (S, HEADS), block: 128 (one thread per d)
// =====================================================================
__global__ void rmsrope(const float* __restrict__ pre, __bf16* __restrict__ out,
                        const float* __restrict__ w_txt, const float* __restrict__ w_img,
                        const float* __restrict__ cosT, const float* __restrict__ sinT,
                        int S, int Stxt, float scale, int do_rope) {
  const int s = blockIdx.x, h = blockIdx.y, d = threadIdx.x;
  __shared__ float buf[HD];
  __shared__ float red[4];

  const float x = pre[(size_t)s * HS + h * HD + d];
  float v = x * x;
#pragma unroll
  for (int off = 16; off > 0; off >>= 1) v += __shfl_down(v, off, 32);
  if ((threadIdx.x & 31) == 0) red[threadIdx.x >> 5] = v;
  __syncthreads();
  const float ssum = red[0] + red[1] + red[2] + red[3];
  const float inv = rsqrtf(ssum * (1.0f / HD) + 1e-6f);
  const float* w = (s < Stxt) ? w_txt : w_img;
  const float xn = x * inv * w[d];
  buf[d] = xn;
  __syncthreads();

  float o = xn;
  if (do_rope) {
    const float c  = cosT[(size_t)s * HD + d];
    const float sn = sinT[(size_t)s * HD + d];
    const float xr = (d & 1) ? buf[d - 1] : -buf[d + 1];
    o = xn * c + xr * sn;
  }
  out[((size_t)h * S + s) * HD + d] = (__bf16)(o * scale);
}

// V reshape/transpose: vpre f32 [S][HS] -> Vt bf16 [H][D][S]
__global__ void prep_v(const float* __restrict__ vpre, __bf16* __restrict__ Vt, int S) {
  const size_t idx = (size_t)blockIdx.x * blockDim.x + threadIdx.x;
  if (idx >= (size_t)S * HS) return;
  const int s = (int)(idx / HS);
  const int c = (int)(idx % HS);
  const int h = c >> 7, d = c & 127;
  Vt[((size_t)(h * HD + d)) * S + s] = (__bf16)vpre[idx];
}

// =====================================================================
// Flash attention: one wave per (16-row Q tile, head).
// Q,K bf16 [H][S][D] (Q pre-scaled by 1/sqrt(D)), Vt bf16 [H][D][S].
// K/V tiles staged into LDS via GLOBAL_LOAD_ASYNC_TO_LDS_B128 when available.
// Output O bf16 [S][H*D].
// =====================================================================
__global__ void attn(const __bf16* __restrict__ Q, const __bf16* __restrict__ K,
                     const __bf16* __restrict__ Vt, __bf16* __restrict__ O, int S) {
  const int lane = threadIdx.x & 31;
  const int half = lane >> 4;
  const int r    = lane & 15;
  const int q0 = blockIdx.x * 16;
  const int h  = blockIdx.y;

  const __bf16* Qh = Q + (size_t)h * S * HD;
  const __bf16* Kh = K + (size_t)h * S * HD;
  const __bf16* Vh = Vt + (size_t)h * HD * S;

  // Q fragments for all 4 K-dim steps (D=128 -> 4 x 32)
  v16bf aq[4];
  const __bf16* qrow = Qh + (size_t)(q0 + r) * HD;
#pragma unroll
  for (int ks = 0; ks < 4; ++ks) {
    const int c0 = ks * 32 + half * 16;
#pragma unroll
    for (int j = 0; j < 8; ++j) {
      aq[ks][j]     = qrow[c0 + j];
      aq[ks][8 + j] = qrow[c0 + 8 + j];
    }
  }

  v8f oacc[8] = {};
  __shared__ float Sbuf[16][32];
  __shared__ float Pbuf[16][32];
  __shared__ float mrow[16], lrow[16], arow[16];
#if HAVE_ASYNC_G2L
  __shared__ __bf16 Ktile[32 * HD];   // 8 KB: 32 keys x 128 dims
  __shared__ __bf16 Vtile[HD * 32];   // 8 KB: 128 dims x 32 keys
#endif
  if (lane < 16) { mrow[lane] = -1e30f; lrow[lane] = 0.0f; }
  __syncthreads();

  for (int kb = 0; kb < S; kb += 32) {
#if HAVE_ASYNC_G2L
    {
      // K tile: rows kb..kb+31 are 8 KB contiguous; 16 x (32 lanes x 16B)
      const char* ksrc = (const char*)(Kh + (size_t)kb * HD);
      char* kdst = (char*)Ktile;
#pragma unroll
      for (int t = 0; t < 16; ++t) {
        const int off = t * 512 + lane * 16;
        async_g2l_b128(ksrc + off, kdst + off);
      }
      // V tile: 128 rows of 64B (stride S*2B); 4 lanes per row
#pragma unroll
      for (int t = 0; t < 16; ++t) {
        const int d  = t * 8 + (lane >> 2);
        const int co = (lane & 3) * 16;
        async_g2l_b128((const char*)(Vh + (size_t)d * S + kb) + co,
                       (char*)(Vtile + d * 32) + co);
      }
      wait_async0();
    }
    __syncthreads();
#endif

    // ---- S = Q @ K^T for 32 keys (2 N-tiles x 4 K-steps = 8 wmma) ----
    v8f sacc[2] = {};
#pragma unroll
    for (int jt = 0; jt < 2; ++jt) {
#if HAVE_ASYNC_G2L
      const __bf16* krow = Ktile + (jt * 16 + r) * HD;
#else
      const __bf16* krow = Kh + (size_t)(kb + jt * 16 + r) * HD;
#endif
#pragma unroll
      for (int ks = 0; ks < 4; ++ks) {
        v16bf bk;
        const int c0 = ks * 32 + half * 16;
#pragma unroll
        for (int j = 0; j < 16; ++j) bk[j] = krow[c0 + j];
        sacc[jt] = wmma_bf16(aq[ks], bk, sacc[jt]);
      }
    }
#pragma unroll
    for (int jt = 0; jt < 2; ++jt)
#pragma unroll
      for (int rr = 0; rr < 8; ++rr)
        Sbuf[rr + half * 8][jt * 16 + r] = sacc[jt][rr];
    __syncthreads();

    // ---- online softmax (lanes 0..15, one row each) ----
    if (lane < 16) {
      float mx = mrow[lane];
#pragma unroll
      for (int j = 0; j < 32; ++j) mx = fmaxf(mx, Sbuf[lane][j]);
      const float al = __expf(mrow[lane] - mx);
      float ls = lrow[lane] * al;
#pragma unroll
      for (int j = 0; j < 32; ++j) {
        const float p = __expf(Sbuf[lane][j] - mx);
        Pbuf[lane][j] = p;
        ls += p;
      }
      mrow[lane] = mx; lrow[lane] = ls; arow[lane] = al;
    }
    __syncthreads();

    // rescale accumulators by per-row alpha
#pragma unroll
    for (int f = 0; f < 8; ++f)
#pragma unroll
      for (int rr = 0; rr < 8; ++rr)
        oacc[f][rr] *= arow[half * 8 + rr];

    // ---- O += P @ V  (P 16x32 as A-frag, 8 D-tiles of V = 8 wmma) ----
    v16bf ap;
    {
      const int c0 = half * 16;
#pragma unroll
      for (int j = 0; j < 8; ++j) {
        ap[j]     = (__bf16)Pbuf[r][c0 + j];
        ap[8 + j] = (__bf16)Pbuf[r][c0 + 8 + j];
      }
    }
#pragma unroll
    for (int f = 0; f < 8; ++f) {
      v16bf bv;
#if HAVE_ASYNC_G2L
      const __bf16* vrow = Vtile + (f * 16 + r) * 32 + half * 16;
#else
      const __bf16* vrow = Vh + (size_t)(f * 16 + r) * S + kb + half * 16;
#endif
#pragma unroll
      for (int j = 0; j < 16; ++j) bv[j] = vrow[j];
      oacc[f] = wmma_bf16(ap, bv, oacc[f]);
    }
    __syncthreads();   // protect LDS tiles / Sbuf / Pbuf for next iteration
  }

  // ---- finalize: divide by row sum, store bf16 [s][h*128 + n] ----
#pragma unroll
  for (int f = 0; f < 8; ++f)
#pragma unroll
    for (int rr = 0; rr < 8; ++rr) {
      const int m = half * 8 + rr;
      const float val = oacc[f][rr] / lrow[m];
      O[(size_t)(q0 + m) * HS + h * HD + f * 16 + r] = (__bf16)val;
    }
}

// f32 [NTOK][HS] -> bf16 [H][NTOK][D] for IP K/V
__global__ void prep_ip(const float* __restrict__ kpre, const float* __restrict__ vpre,
                        __bf16* __restrict__ Kb, __bf16* __restrict__ Vb) {
  const int idx = blockIdx.x * blockDim.x + threadIdx.x;
  if (idx >= NTOK * HS) return;
  const int j = idx / HS;
  const int c = idx % HS;
  const int h = c >> 7, d = c & 127;
  Kb[((size_t)h * NTOK + j) * HD + d] = (__bf16)kpre[idx];
  Vb[((size_t)h * NTOK + j) * HD + d] = (__bf16)vpre[idx];
}

// tiny 4-key IP cross-attention (VALU; ~0.1 GFLOP)
__global__ void ip_attn(const __bf16* __restrict__ Q2, const __bf16* __restrict__ Kb,
                        const __bf16* __restrict__ Vb, float* __restrict__ out, int S) {
  const int idx = blockIdx.x * blockDim.x + threadIdx.x;
  if (idx >= S * HEADS) return;
  const int h = idx % HEADS;
  const int s = idx / HEADS;
  const __bf16* q  = Q2 + ((size_t)h * S + s) * HD;   // already scaled by 1/sqrt(D)
  const __bf16* kk = Kb + (size_t)h * NTOK * HD;
  const __bf16* vv = Vb + (size_t)h * NTOK * HD;

  float sc[NTOK] = {0.f, 0.f, 0.f, 0.f};
  for (int d = 0; d < HD; ++d) {
    const float qv = (float)q[d];
#pragma unroll
    for (int j = 0; j < NTOK; ++j) sc[j] += qv * (float)kk[j * HD + d];
  }
  float mx = sc[0];
#pragma unroll
  for (int j = 1; j < NTOK; ++j) mx = fmaxf(mx, sc[j]);
  float sum = 0.f;
  float p[NTOK];
#pragma unroll
  for (int j = 0; j < NTOK; ++j) { p[j] = __expf(sc[j] - mx); sum += p[j]; }
  const float inv = 1.0f / sum;
  for (int d = 0; d < HD; ++d) {
    float o = 0.f;
#pragma unroll
    for (int j = 0; j < NTOK; ++j) o += p[j] * (float)vv[j * HD + d];
    out[(size_t)s * HS + h * HD + d] = o * inv;
  }
}

// =====================================================================
extern "C" void kernel_launch(void* const* d_in, const int* in_sizes, int n_in,
                              void* d_out, int out_size, void* d_ws, size_t ws_size,
                              hipStream_t stream) {
  (void)in_sizes; (void)n_in; (void)out_size; (void)ws_size;
  const float* hid  = (const float*)d_in[0];
  const float* enc  = (const float*)d_in[1];
  const float* iph  = (const float*)d_in[2];
  const float* cosT = (const float*)d_in[3];
  const float* sinT = (const float*)d_in[4];
  const float *Wq = (const float*)d_in[5],  *bq = (const float*)d_in[6];
  const float *Wk = (const float*)d_in[7],  *bk = (const float*)d_in[8];
  const float *Wv = (const float*)d_in[9],  *bv = (const float*)d_in[10];
  const float *Waq = (const float*)d_in[11], *baq = (const float*)d_in[12];
  const float *Wak = (const float*)d_in[13], *bak = (const float*)d_in[14];
  const float *Wav = (const float*)d_in[15], *bav = (const float*)d_in[16];
  const float *nq = (const float*)d_in[17], *nk = (const float*)d_in[18];
  const float *naq = (const float*)d_in[19], *nak = (const float*)d_in[20];
  const float *Wo = (const float*)d_in[21], *bo = (const float*)d_in[22];
  const float *Wao = (const float*)d_in[23], *bao = (const float*)d_in[24];
  const float *Wkip = (const float*)d_in[25], *bkip = (const float*)d_in[26];
  const float *Wvip = (const float*)d_in[27], *bvip = (const float*)d_in[28];

  // ---------------- workspace layout ----------------
  char* ws = (char*)d_ws;
  const size_t PRE_BYTES = (size_t)STOT * HS * sizeof(float);   // 31,457,280
  const size_t BHD_BYTES = (size_t)HEADS * STOT * HD * 2;       // 15,728,640
  float*  qpre = (float*)(ws);
  float*  kpre = (float*)(ws + PRE_BYTES);
  float*  vpre = (float*)(ws + 2 * PRE_BYTES);
  __bf16* Qb   = (__bf16*)(ws + 3 * PRE_BYTES);
  __bf16* Kb   = (__bf16*)(ws + 3 * PRE_BYTES + BHD_BYTES);
  __bf16* Vt   = (__bf16*)(ws + 3 * PRE_BYTES + 2 * BHD_BYTES);
  __bf16* Ob   = (__bf16*)(ws + 3 * PRE_BYTES + 3 * BHD_BYTES); // STOT*HS bf16
  // post-attention reuse:
  float*  q2pre = qpre;                 // 2048*3072 f32 fits in qpre
  __bf16* Q2b   = (__bf16*)kpre;        // [H][S_IMG][D] bf16 fits in kpre
  float*  ipkpre = vpre;                // [4][3072] f32
  float*  ipvpre = vpre + NTOK * HS;
  __bf16* ipKb = (__bf16*)((char*)vpre + 2 * (size_t)NTOK * HS * sizeof(float));
  __bf16* ipVb = ipKb + NTOK * HS;

  float* out_hs  = (float*)d_out;                         // [2048][3072]
  float* out_enc = out_hs + (size_t)S_IMG * HS;           // [512][3072]
  float* out_ip  = out_enc + (size_t)S_TXT * HS;          // [2048][3072]

  const dim3 blk32(32), blk128(128), blk256(256);
  const dim3 gImg(HS / 64, S_IMG / 32);    // 48 x 64
  const dim3 gTxt(HS / 64, S_TXT / 32);    // 48 x 16
  const dim3 gIp(HS / 64, 1);

  // 1) QKV projections: text rows 0..511, image rows 512..2559
  gemm32<float><<<gImg, blk32, 0, stream>>>(hid, Wq, bq, qpre + (size_t)S_TXT * HS, S_IMG, HS, HS, HS, HS);
  gemm32<float><<<gImg, blk32, 0, stream>>>(hid, Wk, bk, kpre + (size_t)S_TXT * HS, S_IMG, HS, HS, HS, HS);
  gemm32<float><<<gImg, blk32, 0, stream>>>(hid, Wv, bv, vpre + (size_t)S_TXT * HS, S_IMG, HS, HS, HS, HS);
  gemm32<float><<<gTxt, blk32, 0, stream>>>(enc, Waq, baq, qpre, S_TXT, HS, HS, HS, HS);
  gemm32<float><<<gTxt, blk32, 0, stream>>>(enc, Wak, bak, kpre, S_TXT, HS, HS, HS, HS);
  gemm32<float><<<gTxt, blk32, 0, stream>>>(enc, Wav, bav, vpre, S_TXT, HS, HS, HS, HS);

  // 2) RMSNorm + RoPE (+1/sqrt(D) folded into Q); V transpose
  rmsrope<<<dim3(STOT, HEADS), blk128, 0, stream>>>(qpre, Qb, naq, nq, cosT, sinT, STOT, S_TXT, RSQRT_D, 1);
  rmsrope<<<dim3(STOT, HEADS), blk128, 0, stream>>>(kpre, Kb, nak, nk, cosT, sinT, STOT, S_TXT, 1.0f, 1);
  prep_v<<<dim3((unsigned)(((size_t)STOT * HS + 255) / 256)), blk256, 0, stream>>>(vpre, Vt, STOT);

  // 3) joint attention
  attn<<<dim3(STOT / 16, HEADS), blk32, 0, stream>>>(Qb, Kb, Vt, Ob, STOT);

  // 4) output projections
  gemm32<__bf16><<<gImg, blk32, 0, stream>>>(Ob + (size_t)S_TXT * HS, Wo, bo, out_hs, S_IMG, HS, HS, HS, HS);
  gemm32<__bf16><<<gTxt, blk32, 0, stream>>>(Ob, Wao, bao, out_enc, S_TXT, HS, HS, HS, HS);

  // 5) IP-adapter branch
  gemm32<float><<<gImg, blk32, 0, stream>>>(out_hs, Wq, bq, q2pre, S_IMG, HS, HS, HS, HS);
  rmsrope<<<dim3(S_IMG, HEADS), blk128, 0, stream>>>(q2pre, Q2b, nq, nq, cosT, sinT, S_IMG, 0, RSQRT_D, 0);
  gemm32<float><<<gIp, blk32, 0, stream>>>(iph, Wkip, bkip, ipkpre, NTOK, HS, CA, CA, HS);
  gemm32<float><<<gIp, blk32, 0, stream>>>(iph, Wvip, bvip, ipvpre, NTOK, HS, CA, CA, HS);
  prep_ip<<<dim3((NTOK * HS + 255) / 256), blk256, 0, stream>>>(ipkpre, ipvpre, ipKb, ipVb);
  ip_attn<<<dim3((S_IMG * HEADS + 255) / 256), blk256, 0, stream>>>(Q2b, ipKb, ipVb, out_ip, S_IMG);
}
